// MLA_8933531976382
// MI455X (gfx1250) — matrix operand order repaired and checked by reading, not compiled, not run on hardware
//
#include <hip/hip_runtime.h>
#include <hip/hip_bf16.h>

typedef unsigned short u16;
typedef unsigned int   u32;
typedef unsigned long long u64;
typedef __bf16 v16bf __attribute__((ext_vector_type(16)));
typedef float  v8f   __attribute__((ext_vector_type(8)));
typedef unsigned int v4u __attribute__((ext_vector_type(4)));
typedef int v8i __attribute__((ext_vector_type(8)));
typedef int v4i __attribute__((ext_vector_type(4)));

#define D_MODEL 3072
#define N_HEADS 24
#define SEQ     2048
#define BATCH   2
#define DH      128
#define Q_PROJ  1536
#define KV_PROJ 2048
#define ROPE    64
#define NOPE    64
#define MROWS   (BATCH*SEQ)              // 4096
#define KV_OUT  (KV_PROJ + ROPE)         // 2112
#define UKV_OUT (D_MODEL + N_HEADS*NOPE) // 4608

#if defined(__has_builtin)
#if __has_builtin(__builtin_amdgcn_tensor_load_to_lds) && __has_builtin(__builtin_amdgcn_s_wait_tensorcnt)
#define HAVE_TDM 1
#endif
#endif

// ---------- bf16 helpers ----------
__device__ __forceinline__ u16 f2bf(float x) {
    union { float f; u32 u; } v; v.f = x;
    u32 r = v.u + 0x7FFFu + ((v.u >> 16) & 1u);
    return (u16)(r >> 16);
}

__device__ __forceinline__ v8f zero8() {
    v8f z;
#pragma unroll
    for (int i = 0; i < 8; ++i) z[i] = 0.0f;
    return z;
}

// A-fragment (16x32 bf16): lane = row m (lane&15), k = {base..base+7, base+16..base+23},
// base = (lane>>4)*8.  p -> &buf[row*stride + kc + (lane>>4)*8]
__device__ __forceinline__ v16bf ld_frag_a(const u16* p) {
    union { uint4 u[2]; v16bf v; } u;
    u.u[0] = *reinterpret_cast<const uint4*>(p);
    u.u[1] = *reinterpret_cast<const uint4*>(p + 16);
    return u.v;
}

// B-fragment (32x16 bf16) from B^T staged [n][k]: lane = col n (lane&15),
// k = {base..base+7, base+16..base+23}, base = (lane>>4)*8.  p -> &bufT[n*stride + kc + (lane>>4)*8]
__device__ __forceinline__ v16bf ld_frag_b(const u16* p) {
    union { uint4 u[2]; v16bf v; } u;
    u.u[0] = *reinterpret_cast<const uint4*>(p);
    u.u[1] = *reinterpret_cast<const uint4*>(p + 16);
    return u.v;
}

#ifdef HAVE_TDM
// ---------- Tensor Data Mover: 2-D tile (2-byte elems) global -> LDS ----------
// Loads tile [tile1 rows x tile0 elems], row stride = stride0 elems, linear into LDS
// (row-major [tile1][tile0]). Rows y >= tdim1 are zero-filled by the TDM OOB rule.
__device__ __forceinline__ void tdm_load_2d(u32 lds_addr, const void* gptr,
                                            u32 tile0, u32 tile1,
                                            u64 stride0, u32 tdim1) {
    u64 ga = (u64)gptr;
    v4u g0;
    g0[0] = 1u;                                            // count=1 (valid user D#)
    g0[1] = lds_addr;                                      // LDS byte offset
    g0[2] = (u32)ga;                                       // global_addr[31:0]
    g0[3] = (u32)((ga >> 32) & 0x1FFFFFFu) | (2u << 30);   // global_addr[56:32] | type=2
    u32 td0 = tile0;                                       // dim0 always fully in-bounds
    v8i g1;
    g1[0] = (int)(1u << 16);                               // data_size=1 -> 2 bytes
    g1[1] = (int)((td0 & 0xFFFFu) << 16);                  // tensor_dim0[15:0]
    g1[2] = (int)(((td0 >> 16) & 0xFFFFu) | ((tdim1 & 0xFFFFu) << 16));
    g1[3] = (int)(((tdim1 >> 16) & 0xFFFFu) | ((tile0 & 0xFFFFu) << 16)); // tile_dim0
    g1[4] = (int)(tile1 & 0xFFFFu);                        // tile_dim1 (tile_dim2=0)
    g1[5] = (int)(u32)(stride0 & 0xFFFFFFFFull);           // tensor_dim0_stride[31:0]
    g1[6] = (int)(u32)((stride0 >> 32) & 0xFFFFull);       // stride[47:32]
    g1[7] = 0;
    v4i gz = {0, 0, 0, 0};
#if __clang_major__ >= 23
    v8i gz8 = {0, 0, 0, 0, 0, 0, 0, 0};
    __builtin_amdgcn_tensor_load_to_lds(g0, g1, gz, gz, gz8, 0);
#else
    __builtin_amdgcn_tensor_load_to_lds(g0, g1, gz, gz, 0);
#endif
}
__device__ __forceinline__ u32 lds_off(const void* p) { return (u32)(u64)p; }
#endif

// ---------- conversion kernels ----------
__global__ void cvt_f32_bf16(const float* __restrict__ in, u16* __restrict__ out, int n) {
    int i = blockIdx.x * blockDim.x + threadIdx.x;
    if (i < n) out[i] = f2bf(in[i]);
}

// out[c*rows + r] = bf16(in[r*cols + c])  (build W^T so GEMM B operands are [N,K] row-major)
__global__ void cvtT_f32_bf16(const float* __restrict__ in, u16* __restrict__ out, int rows, int cols) {
    int i = blockIdx.x * blockDim.x + threadIdx.x;
    if (i >= rows * cols) return;
    int r = i / cols, c = i % cols;
    out[(size_t)c * rows + r] = f2bf(in[i]);
}

// ---------- bf16 GEMM: C[M,N] f32 = A[M,K] x B[K,N] with B given transposed Bt[N,K].
// Block tile 128x256, K-step 32, 8 waves x (64x64) -> 16 WMMAs per wave per step.
// TDM double-buffered staging (wave 0 drives the DMA), manual vector fallback otherwise.
__global__ __launch_bounds__(256) void gemm_bf16(const u16* __restrict__ A,
                                                 const u16* __restrict__ Bt,
                                                 float* __restrict__ C,
                                                 int M, int N, int K) {
    __shared__ __align__(16) u16 ldsA[2][128 * 32];
    __shared__ __align__(16) u16 ldsB[2][256 * 32];   // [n][k]

    const int t = threadIdx.x;
    const int lane = t & 31, w = t >> 5;
    const int wm = w >> 2, wn = w & 3;                // 2x4 wave grid (64x64 tiles)
    const int bm = blockIdx.y * 128, bn = blockIdx.x * 256;
    const int ln = lane & 15, hl = lane >> 4;
    const int niter = K / 32;

    v8f acc[4][4];
#pragma unroll
    for (int i = 0; i < 4; ++i)
#pragma unroll
        for (int j = 0; j < 4; ++j) acc[i][j] = zero8();

#ifdef HAVE_TDM
    const u32 nrem = (u32)(N - bn);                   // rows of Bt tile in-bounds (rest zero-fill)
    if (w == 0) {
        tdm_load_2d(lds_off(&ldsA[0][0]), A  + (size_t)bm * K, 32, 128, (u64)K, 128u);
        tdm_load_2d(lds_off(&ldsB[0][0]), Bt + (size_t)bn * K, 32, 256, (u64)K, nrem);
    }
    for (int it = 0; it < niter; ++it) {
        const int buf = it & 1;
        if (w == 0) {
            if (it + 1 < niter) {
                const int k0 = (it + 1) * 32;
                tdm_load_2d(lds_off(&ldsA[buf ^ 1][0]), A  + (size_t)bm * K + k0, 32, 128, (u64)K, 128u);
                tdm_load_2d(lds_off(&ldsB[buf ^ 1][0]), Bt + (size_t)bn * K + k0, 32, 256, (u64)K, nrem);
                __builtin_amdgcn_s_wait_tensorcnt(2);  // older pair (current tile) done
            } else {
                __builtin_amdgcn_s_wait_tensorcnt(0);
            }
        }
        __syncthreads();
        const u16* La = &ldsA[buf][0];
        const u16* Lb = &ldsB[buf][0];
        v16bf af[4];
#pragma unroll
        for (int i = 0; i < 4; ++i)
            af[i] = ld_frag_a(&La[(size_t)(wm * 64 + i * 16 + ln) * 32 + hl * 8]);
#pragma unroll
        for (int j = 0; j < 4; ++j) {
            v16bf bfv = ld_frag_b(&Lb[(size_t)(wn * 64 + j * 16 + ln) * 32 + hl * 8]);
#pragma unroll
            for (int i = 0; i < 4; ++i)
                acc[i][j] = __builtin_amdgcn_wmma_f32_16x16x32_bf16(
                    false, af[i], false, bfv, (short)0, acc[i][j], false, false);
        }
        __syncthreads();
    }
#else
    for (int it = 0; it < niter; ++it) {
        const int k0 = it * 32;
        // Stage A tile [128][32]: 512 x uint4 / 256 threads
#pragma unroll
        for (int j = 0; j < 2; ++j) {
            int i = t + 256 * j;
            int r = i >> 2, c = (i & 3) * 8;
            *reinterpret_cast<uint4*>(&ldsA[0][r * 32 + c]) =
                *reinterpret_cast<const uint4*>(&A[(size_t)(bm + r) * K + k0 + c]);
        }
        // Stage Bt tile [256][32]: 1024 x uint4 / 256 threads (row-guarded)
#pragma unroll
        for (int j = 0; j < 4; ++j) {
            int i = t + 256 * j;
            int n = i >> 2, c = (i & 3) * 8;
            uint4 v = {0, 0, 0, 0};
            if (bn + n < N)
                v = *reinterpret_cast<const uint4*>(&Bt[(size_t)(bn + n) * K + k0 + c]);
            *reinterpret_cast<uint4*>(&ldsB[0][n * 32 + c]) = v;
        }
        __syncthreads();
        const u16* La = &ldsA[0][0];
        const u16* Lb = &ldsB[0][0];
        v16bf af[4];
#pragma unroll
        for (int i = 0; i < 4; ++i)
            af[i] = ld_frag_a(&La[(size_t)(wm * 64 + i * 16 + ln) * 32 + hl * 8]);
#pragma unroll
        for (int j = 0; j < 4; ++j) {
            v16bf bfv = ld_frag_b(&Lb[(size_t)(wn * 64 + j * 16 + ln) * 32 + hl * 8]);
#pragma unroll
            for (int i = 0; i < 4; ++i)
                acc[i][j] = __builtin_amdgcn_wmma_f32_16x16x32_bf16(
                    false, af[i], false, bfv, (short)0, acc[i][j], false, false);
        }
        __syncthreads();
    }
#endif

    // Epilogue: C layout -> lane: n=lane&15, vgpr r: m = r + 8*(lane>=16)
#pragma unroll
    for (int i = 0; i < 4; ++i)
#pragma unroll
        for (int j = 0; j < 4; ++j) {
            int col = bn + wn * 64 + j * 16 + ln;
            if (col < N) {
#pragma unroll
                for (int r = 0; r < 8; ++r) {
                    int row = bm + wm * 64 + i * 16 + r + 8 * hl;
                    C[(size_t)row * N + col] = acc[i][j][r];
                }
            }
        }
}

// ---------- layernorm: one row per block, f32 in -> bf16 out ----------
__global__ __launch_bounds__(256) void ln_kernel(const float* __restrict__ in, int istride, int L,
                                                 const float* __restrict__ w, const float* __restrict__ bias,
                                                 u16* __restrict__ out, int ostride) {
    __shared__ float red[256];
    const int row = blockIdx.x, t = threadIdx.x;
    const float* x = in + (size_t)row * istride;

    float s = 0.f;
    for (int i = t; i < L; i += 256) s += x[i];
    red[t] = s; __syncthreads();
    for (int st = 128; st > 0; st >>= 1) { if (t < st) red[t] += red[t + st]; __syncthreads(); }
    float mean = red[0] / (float)L;
    __syncthreads();

    float v = 0.f;
    for (int i = t; i < L; i += 256) { float d = x[i] - mean; v += d * d; }
    red[t] = v; __syncthreads();
    for (int st = 128; st > 0; st >>= 1) { if (t < st) red[t] += red[t + st]; __syncthreads(); }
    float rstd = rsqrtf(red[0] / (float)L + 1e-5f);

    u16* o = out + (size_t)row * ostride;
    for (int i = t; i < L; i += 256)
        o[i] = f2bf((x[i] - mean) * rstd * w[i] + bias[i]);
}

// ---------- RoPE ----------
__device__ __forceinline__ float rope_val(const float* base, int j, int s) {
    int fi = j & 31;                                   // freq_i = 10000^(-i/64)
    float freq = __expf(-(float)fi * (9.210340371976184f / 64.0f));
    float ang = (float)s * freq;
    float x  = base[j];
    float xr = (j < 32) ? -base[j + 32] : base[j - 32];
    return x * __cosf(ang) + xr * __sinf(ang);
}

// Qh [B,H,S,128] bf16 from Q f32 [B,S,3072] (rope on last 64 of each head)
__global__ void prep_q_kernel(const float* __restrict__ Qf, u16* __restrict__ Qh) {
    int idx = blockIdx.x * blockDim.x + threadIdx.x;
    if (idx >= BATCH * N_HEADS * SEQ * DH) return;
    int d = idx & 127;
    int s = (idx >> 7) & (SEQ - 1);
    int hb = idx >> 18;                                // == b*24 + h
    int h = hb % N_HEADS;
    int b = hb / N_HEADS;
    const float* q = Qf + ((size_t)(b * SEQ + s)) * D_MODEL + h * DH;
    float val = (d < NOPE) ? q[d] : rope_val(q + NOPE, d - NOPE, s);
    Qh[idx] = f2bf(val);
}

// Kh [B,H,S,128] bf16 and Vt [B,H,128,S] bf16 (V pre-transposed for attention staging)
__global__ void prep_kv_kernel(const float* __restrict__ KVf, const float* __restrict__ ckv,
                               u16* __restrict__ Kh, u16* __restrict__ Vt) {
    int idx = blockIdx.x * blockDim.x + threadIdx.x;
    if (idx >= BATCH * N_HEADS * SEQ * DH) return;
    int d = idx & 127;
    int s = (idx >> 7) & (SEQ - 1);
    int hb = idx >> 18;                                // == b*24 + h
    int h = hb % N_HEADS;
    int b = hb / N_HEADS;
    const float* kv = KVf + ((size_t)(b * SEQ + s)) * UKV_OUT + h * (DH + NOPE);
    float kval;
    if (d < NOPE) {
        kval = kv[d];
    } else {
        const float* kr = ckv + ((size_t)(b * SEQ + s)) * KV_OUT + KV_PROJ;
        kval = rope_val(kr, d - NOPE, s);
    }
    Kh[idx] = f2bf(kval);
    Vt[((size_t)hb * DH + d) * SEQ + s] = f2bf(kv[NOPE + d]);
}

// ---------- flash attention: block = 128 q rows (8 waves x 16), 32-key tiles ----------
__global__ __launch_bounds__(256) void attn_kernel(const u16* __restrict__ Qh,
                                                   const u16* __restrict__ Kh,
                                                   const u16* __restrict__ Vt,
                                                   u16* __restrict__ attn_out) {
    __shared__ __align__(16) u16 ldsK[32 * 128];      // [key][d]  (B^T for Q.K^T)
    __shared__ __align__(16) u16 ldsVt[128 * 32];     // [d][key]  (B^T for P.V)
    __shared__ __align__(16) u16 ldsP[8][16 * 32];    // per-wave P scratch [m][key]

    const int t = threadIdx.x, lane = t & 31, w = t >> 5;
    const int ln = lane & 15, hl = lane >> 4;
    const int qb = blockIdx.x * 128;
    const int h = blockIdx.y, b = blockIdx.z;
    const int qw = qb + w * 16;

    const size_t hoff = ((size_t)(b * N_HEADS + h)) * SEQ * DH;
    const u16* Qp  = Qh + hoff;
    const u16* Kp  = Kh + hoff;
    const u16* Vtp = Vt + hoff;                       // [128][SEQ]

    // Preload Q fragments (16 rows x 128 d = 4 chunks of 32)
    v16bf qf[4];
    {
        const u16* qrow = Qp + (size_t)(qw + ln) * DH;
#pragma unroll
        for (int c = 0; c < 4; ++c) {
            int base = c * 32 + hl * 8;
            union { uint4 u[2]; v16bf v; } u;
            u.u[0] = *reinterpret_cast<const uint4*>(qrow + base);
            u.u[1] = *reinterpret_cast<const uint4*>(qrow + base + 16);
            qf[c] = u.v;
        }
    }

    v8f o[8];
#pragma unroll
    for (int c = 0; c < 8; ++c) o[c] = zero8();
    float m[8], l[8];
#pragma unroll
    for (int r = 0; r < 8; ++r) { m[r] = -1e30f; l[r] = 0.0f; }
    const float scale = 0.08838834764831845f;          // 1/sqrt(128)

    const int niter = (qb + 128) / 32;
    for (int it = 0; it < niter; ++it) {
        const int kb = it * 32;
#ifdef HAVE_TDM
        if (w == 0) {
            tdm_load_2d(lds_off(&ldsK[0]),  Kp + (size_t)kb * DH, 128, 32, (u64)DH, 32u);
            tdm_load_2d(lds_off(&ldsVt[0]), Vtp + kb,              32, 128, (u64)SEQ, 128u);
            __builtin_amdgcn_s_wait_tensorcnt(0);
        }
#else
        // K tile [32][128]
#pragma unroll
        for (int j = 0; j < 2; ++j) {
            int i = t + 256 * j;
            int key = i >> 4, c = (i & 15) * 8;
            *reinterpret_cast<uint4*>(&ldsK[key * 128 + c]) =
                *reinterpret_cast<const uint4*>(&Kp[(size_t)(kb + key) * DH + c]);
        }
        // V^T tile [128][32] (contiguous rows thanks to Vt layout)
#pragma unroll
        for (int j = 0; j < 2; ++j) {
            int i = t + 256 * j;
            int d = i >> 2, c = (i & 3) * 8;
            *reinterpret_cast<uint4*>(&ldsVt[d * 32 + c]) =
                *reinterpret_cast<const uint4*>(&Vtp[(size_t)d * SEQ + kb + c]);
        }
#endif
        __syncthreads();

        // Scores: two 16x16 tiles covering 32 keys
        v8f s0a = zero8(), s1a = zero8();
#pragma unroll
        for (int c = 0; c < 4; ++c) {
            v16bf kf0 = ld_frag_b(&ldsK[(size_t)(0 * 16 + ln) * 128 + c * 32 + hl * 8]);
            s0a = __builtin_amdgcn_wmma_f32_16x16x32_bf16(false, qf[c], false, kf0, (short)0, s0a, false, false);
            v16bf kf1 = ld_frag_b(&ldsK[(size_t)(1 * 16 + ln) * 128 + c * 32 + hl * 8]);
            s1a = __builtin_amdgcn_wmma_f32_16x16x32_bf16(false, qf[c], false, kf1, (short)0, s1a, false, false);
        }

        // Online softmax (row m lives in one 16-lane half; xor-butterfly row reduce)
        float p0[8], p1[8];
#pragma unroll
        for (int r = 0; r < 8; ++r) {
            int qrow = qw + r + 8 * hl;
            float v0 = s0a[r] * scale; if (kb + ln > qrow)      v0 = -1e30f;
            float v1 = s1a[r] * scale; if (kb + 16 + ln > qrow) v1 = -1e30f;
            float tm = fmaxf(v0, v1);
            tm = fmaxf(tm, __shfl_xor(tm, 8, 32));
            tm = fmaxf(tm, __shfl_xor(tm, 4, 32));
            tm = fmaxf(tm, __shfl_xor(tm, 2, 32));
            tm = fmaxf(tm, __shfl_xor(tm, 1, 32));
            float nm = fmaxf(m[r], tm);
            float alpha = __expf(m[r] - nm);
            m[r] = nm;
            p0[r] = __expf(v0 - nm);
            p1[r] = __expf(v1 - nm);
            float rs = p0[r] + p1[r];
            rs += __shfl_xor(rs, 8, 32);
            rs += __shfl_xor(rs, 4, 32);
            rs += __shfl_xor(rs, 2, 32);
            rs += __shfl_xor(rs, 1, 32);
            l[r] = l[r] * alpha + rs;
#pragma unroll
            for (int c = 0; c < 8; ++c) o[c][r] = o[c][r] * alpha;
        }

        // Re-lay-out P into A-fragment form via per-wave LDS scratch
        u16* pp = &ldsP[w][0];
#pragma unroll
        for (int r = 0; r < 8; ++r) {
            int row = r + 8 * hl;
            pp[row * 32 + ln]      = f2bf(p0[r]);
            pp[row * 32 + 16 + ln] = f2bf(p1[r]);
        }
        // same-wave DS ops are in-order; safe to read back
        v16bf pf = ld_frag_a(&pp[(size_t)ln * 32 + hl * 8]);

        // O += P x V  (8 d-chunks of 16)
#pragma unroll
        for (int c = 0; c < 8; ++c) {
            v16bf vf = ld_frag_b(&ldsVt[(size_t)(c * 16 + ln) * 32 + hl * 8]);
            o[c] = __builtin_amdgcn_wmma_f32_16x16x32_bf16(false, pf, false, vf, (short)0, o[c], false, false);
        }
        __syncthreads();
    }

    // Epilogue: normalize and scatter to [B,S, h*128+d] bf16
#pragma unroll
    for (int c = 0; c < 8; ++c) {
        int d = c * 16 + ln;
#pragma unroll
        for (int r = 0; r < 8; ++r) {
            int qrow = qw + r + 8 * hl;
            float v = o[c][r] / l[r];
            attn_out[((size_t)(b * SEQ + qrow)) * D_MODEL + h * DH + d] = f2bf(v);
        }
    }
}

// ---------- host orchestration ----------
extern "C" void kernel_launch(void* const* d_in, const int* in_sizes, int n_in,
                              void* d_out, int out_size, void* d_ws, size_t ws_size,
                              hipStream_t stream) {
    const float* x       = (const float*)d_in[0];
    const float* W_dq    = (const float*)d_in[1];
    const float* W_uq    = (const float*)d_in[2];
    const float* q_ln_w  = (const float*)d_in[3];
    const float* q_ln_b  = (const float*)d_in[4];
    const float* W_dkv   = (const float*)d_in[5];
    const float* W_ukv   = (const float*)d_in[6];
    const float* kv_ln_w = (const float*)d_in[7];
    const float* kv_ln_b = (const float*)d_in[8];
    const float* W_o     = (const float*)d_in[9];

    float* out1 = (float*)d_out;                       // [4096,3072]
    float* ckv  = out1 + (size_t)MROWS * D_MODEL;      // [4096,2112]

    char* ws = (char*)d_ws;
    size_t off = 0;
    auto alloc = [&](size_t bytes) -> void* {
        void* p = ws + off;
        off += (bytes + 255) & ~(size_t)255;
        return p;
    };

    // All GEMM B operands stored pre-transposed: Bt[N][K] row-major bf16.
    u16*   xb     = (u16*)  alloc((size_t)MROWS * D_MODEL * 2);
    u16*   wdqT   = (u16*)  alloc((size_t)D_MODEL * Q_PROJ * 2);   // [1536][3072]
    u16*   wuqT   = (u16*)  alloc((size_t)Q_PROJ * D_MODEL * 2);   // [3072][1536]
    u16*   wdkvT  = (u16*)  alloc((size_t)D_MODEL * KV_OUT * 2);   // [2112][3072]
    u16*   wukvT  = (u16*)  alloc((size_t)KV_PROJ * UKV_OUT * 2);  // [4608][2048]
    u16*   wo_b   = (u16*)  alloc((size_t)D_MODEL * D_MODEL * 2);  // W_o itself == (W_o^T)^T
    float* cqf    = (float*)alloc((size_t)MROWS * Q_PROJ * 4);
    u16*   cqb    = (u16*)  alloc((size_t)MROWS * Q_PROJ * 2);
    float* Qf     = (float*)alloc((size_t)MROWS * D_MODEL * 4);
    u16*   kvb    = (u16*)  alloc((size_t)MROWS * KV_PROJ * 2);
    float* KVf    = (float*)alloc((size_t)MROWS * UKV_OUT * 4);
    u16*   Qh     = (u16*)  alloc((size_t)BATCH * N_HEADS * SEQ * DH * 2);
    u16*   Kh     = (u16*)  alloc((size_t)BATCH * N_HEADS * SEQ * DH * 2);
    u16*   Vt     = (u16*)  alloc((size_t)BATCH * N_HEADS * SEQ * DH * 2);
    u16*   attnb  = (u16*)  alloc((size_t)MROWS * D_MODEL * 2);

    auto cvt = [&](const float* in, u16* out, size_t n) {
        cvt_f32_bf16<<<dim3((unsigned)((n + 255) / 256)), dim3(256), 0, stream>>>(in, out, (int)n);
    };
    auto cvtT = [&](const float* in, u16* out, int rows, int cols) {
        cvtT_f32_bf16<<<dim3((unsigned)(((size_t)rows * cols + 255) / 256)), dim3(256), 0, stream>>>(
            in, out, rows, cols);
    };

    // 1) bf16 conversions (weights transposed so GEMM sees Bt[N][K])
    cvt (x,      xb,    (size_t)MROWS * D_MODEL);
    cvtT(W_dq,   wdqT,  D_MODEL, Q_PROJ);
    cvtT(W_uq,   wuqT,  Q_PROJ,  D_MODEL);
    cvtT(W_dkv,  wdkvT, D_MODEL, KV_OUT);
    cvtT(W_ukv,  wukvT, KV_PROJ, UKV_OUT);
    cvt (W_o,    wo_b,  (size_t)D_MODEL * D_MODEL);    // final GEMM uses W_o^T -> Bt == W_o

    // 2) cq = x @ W_dq ; LN -> bf16
    gemm_bf16<<<dim3(Q_PROJ / 256, MROWS / 128), dim3(256), 0, stream>>>(
        xb, wdqT, cqf, MROWS, Q_PROJ, D_MODEL);
    ln_kernel<<<dim3(MROWS), dim3(256), 0, stream>>>(
        cqf, Q_PROJ, Q_PROJ, q_ln_w, q_ln_b, cqb, Q_PROJ);

    // 3) Q = cq_ln @ W_uq
    gemm_bf16<<<dim3(D_MODEL / 256, MROWS / 128), dim3(256), 0, stream>>>(
        cqb, wuqT, Qf, MROWS, D_MODEL, Q_PROJ);

    // 4) compressed_kv = x @ W_dkv  (written directly into d_out region 2)
    gemm_bf16<<<dim3((KV_OUT + 255) / 256, MROWS / 128), dim3(256), 0, stream>>>(
        xb, wdkvT, ckv, MROWS, KV_OUT, D_MODEL);
    ln_kernel<<<dim3(MROWS), dim3(256), 0, stream>>>(
        ckv, KV_OUT, KV_PROJ, kv_ln_w, kv_ln_b, kvb, KV_PROJ);

    // 5) KV = kv_ln @ W_ukv
    gemm_bf16<<<dim3(UKV_OUT / 256, MROWS / 128), dim3(256), 0, stream>>>(
        kvb, wukvT, KVf, MROWS, UKV_OUT, KV_PROJ);

    // 6) head split + RoPE -> per-head bf16 Q, K and V^T
    {
        int n = BATCH * N_HEADS * SEQ * DH;
        prep_q_kernel<<<dim3((n + 255) / 256), dim3(256), 0, stream>>>(Qf, Qh);
        prep_kv_kernel<<<dim3((n + 255) / 256), dim3(256), 0, stream>>>(KVf, ckv, Kh, Vt);
    }

    // 7) causal flash attention
    attn_kernel<<<dim3(SEQ / 128, N_HEADS, BATCH), dim3(256), 0, stream>>>(
        Qh, Kh, Vt, attnb);

    // 8) out = attn_out @ W_o^T  (written directly into d_out region 1)
    gemm_bf16<<<dim3(D_MODEL / 256, MROWS / 128), dim3(256), 0, stream>>>(
        attnb, wo_b, out1, MROWS, D_MODEL, D_MODEL);

    (void)in_sizes; (void)n_in; (void)out_size; (void)ws_size;
}
